// myModel_88991722373415
// MI455X (gfx1250) — compile-verified, matrix operand order checked by reading, not run on hardware
//
#include <hip/hip_runtime.h>
#include <hip/hip_bf16.h>

// ---------------------------------------------------------------------------
// Types for CDNA5 WMMA
// ---------------------------------------------------------------------------
typedef __attribute__((ext_vector_type(16))) __bf16 bf16x16;
typedef __attribute__((ext_vector_type(8)))  float  f32x8;

union FragU { uint4 u[2]; bf16x16 f; };

#define CB  4096           // batch
#define CU  4              // n_ue
#define CA  64             // BS antennas
#define CT  64             // pilot length
#define CM  (CU*CB)        // 16384 rows for all big GEMMs
#define CD  64             // VQ dim
#define CK  256            // codebook size
#define CNQ 4              // quantizers

#define ROWP 80            // LDS row pitch in bytes (64B payload + 16B pad)

__device__ __forceinline__ unsigned short f2bf(float x) {
    unsigned int u = __float_as_uint(x);
    unsigned int r = (u + 0x7FFFu + ((u >> 16) & 1u)) >> 16;
    return (unsigned short)r;
}

__device__ __forceinline__ float mishf(float x) {
    float sp = (x > 20.f) ? x : log1pf(__expf(x));
    return x * tanhf(sp);
}

// async copy of one 16B chunk: LDS[lds_off] <= MEM[gaddr]  (per-lane, ASYNCcnt)
__device__ __forceinline__ void async_cp16(unsigned lds_off, const void* g) {
    asm volatile("global_load_async_to_lds_b128 %0, %1, off"
                 :: "v"(lds_off), "v"((unsigned long long)(uintptr_t)g)
                 : "memory");
}
__device__ __forceinline__ void wait_async6() {
    asm volatile("s_wait_asynccnt 0x6" ::: "memory");
}
__device__ __forceinline__ void wait_async0() {
    asm volatile("s_wait_asynccnt 0x0" ::: "memory");
}

// ---------------------------------------------------------------------------
// Generic elementwise kernels
// ---------------------------------------------------------------------------
__global__ void k_zero_tail(float* o) {
    if (threadIdx.x < 6) o[2097152 + threadIdx.x] = 0.f;
}

// fp32 [K,N] -> bf16 transposed [N,K]
__global__ void k_f2bf_t(const float* __restrict__ s, unsigned short* __restrict__ d,
                         int N, int K) {
    int i = blockIdx.x * 256 + threadIdx.x;
    if (i >= N * K) return;
    int n = i / K, k = i - n * K;
    d[i] = f2bf(s[(size_t)k * N + n]);
}

__global__ void k_mish_bf(const float* __restrict__ x, unsigned short* __restrict__ o, size_t n) {
    size_t i = (size_t)blockIdx.x * 256 + threadIdx.x;
    if (i < n) o[i] = f2bf(mishf(x[i]));
}

__global__ void k_tanh_f32(const float* __restrict__ x, float* __restrict__ o, size_t n) {
    size_t i = (size_t)blockIdx.x * 256 + threadIdx.x;
    if (i < n) o[i] = tanhf(x[i]);
}

// ---------------------------------------------------------------------------
// Pilot: column norms of Wp, then pilot = h @ Wp (+ noise), emit bf16 [U,B,2T]
// ---------------------------------------------------------------------------
__global__ void k_pilot_norm(const float* __restrict__ Wr, const float* __restrict__ Wi,
                             float* __restrict__ norms) {
    int t = threadIdx.x;  // 64 threads
    float s = 0.f;
    for (int a = 0; a < CA; ++a) {
        float r = Wr[a * CT + t], im = Wi[a * CT + t];
        s += r * r + im * im;
    }
    norms[t] = sqrtf(s);
}

__global__ void k_pilot(const float* __restrict__ xr, const float* __restrict__ xi,
                        const float* __restrict__ Wr, const float* __restrict__ Wi,
                        const float* __restrict__ norms, const float* __restrict__ noise,
                        unsigned short* __restrict__ pcat) {
    int gid = blockIdx.x * 256 + threadIdx.x;       // U*B*T = 1,048,576
    int t  = gid & 63;
    int ub = gid >> 6;
    int u  = ub >> 12;          // / 4096
    int b  = ub & 4095;
    const float* hr = xr + ((size_t)b * CU + u) * CA;
    const float* hi = xi + ((size_t)b * CU + u) * CA;
    float pr = 0.f, pi = 0.f;
    for (int a = 0; a < CA; ++a) {
        float wr = Wr[a * CT + t], wi = Wi[a * CT + t];
        pr += hr[a] * wr - hi[a] * wi;
        pi += hr[a] * wi + hi[a] * wr;
    }
    float inv = 1.f / norms[t];
    float nn  = noise[((size_t)b * CU + u) * CT + t] * 0.22360679775f;  // sqrt(0.1/2)
    pr = pr * inv + nn;
    pi = pi * inv + nn;
    size_t row = (size_t)u * CB + b;
    pcat[row * (2 * CT) + t]      = f2bf(pr);
    pcat[row * (2 * CT) + CT + t] = f2bf(pi);
}

// ---------------------------------------------------------------------------
// WMMA bf16 GEMM: C[M,N] = A[M,K] @ Wt^T + bias
//   A  : [M,K] bf16 row-major, Wt : [N,K] bf16 (pre-transposed weights)
//   block = 128 threads (4 waves, 2x2), block tile 128x64, wave tile 64x32
//   K-step 32, double-buffered LDS filled by GLOBAL_LOAD_ASYNC_TO_LDS_B128
// ---------------------------------------------------------------------------
__device__ __forceinline__ void stage_tiles(const unsigned short* __restrict__ A,
                                            const unsigned short* __restrict__ Wt,
                                            int m0, int n0, int K, int k0,
                                            unsigned aoff, unsigned boff, int tid) {
    // A tile: 128 rows x 32 bf16 = 512 chunks of 16B; 4 per thread
    for (int i = 0; i < 4; ++i) {
        int c = tid + i * 128;
        int row = c >> 2, col = c & 3;
        const unsigned short* g = A + (size_t)(m0 + row) * K + k0 + col * 8;
        async_cp16(aoff + (unsigned)(row * ROWP + col * 16), g);
    }
    // B tile: 64 rows x 32 bf16 = 256 chunks; 2 per thread
    for (int i = 0; i < 2; ++i) {
        int c = tid + i * 128;
        int row = c >> 2, col = c & 3;
        const unsigned short* g = Wt + (size_t)(n0 + row) * K + k0 + col * 8;
        async_cp16(boff + (unsigned)(row * ROWP + col * 16), g);
    }
}

__global__ void k_gemm(const unsigned short* __restrict__ A,
                       const unsigned short* __restrict__ Wt,
                       const float* __restrict__ bias,
                       float* __restrict__ C, int N, int K) {
    __shared__ __align__(16) char lAb[2][128 * ROWP];   // 20 KB
    __shared__ __align__(16) char lBb[2][64 * ROWP];    // 10 KB

    const int tid  = threadIdx.x;
    const int wave = tid >> 5;
    const int lane = tid & 31;
    const int half = lane >> 4;
    const int l15  = lane & 15;
    const int wrow = wave >> 1;     // 0..1  -> 64-row slice
    const int wcol = wave & 1;      // 0..1  -> 32-col slice
    const int m0   = blockIdx.y * 128;
    const int n0   = blockIdx.x * 64;
    const int nk   = K >> 5;

    const unsigned a0 = (unsigned)(uintptr_t)(&lAb[0][0]);
    const unsigned a1 = (unsigned)(uintptr_t)(&lAb[1][0]);
    const unsigned b0 = (unsigned)(uintptr_t)(&lBb[0][0]);
    const unsigned b1 = (unsigned)(uintptr_t)(&lBb[1][0]);

    f32x8 acc[4][2] = {};

    stage_tiles(A, Wt, m0, n0, K, 0, a0, b0, tid);

    for (int it = 0; it < nk; ++it) {
        const int buf = it & 1;
        if (it + 1 < nk) {
            stage_tiles(A, Wt, m0, n0, K, (it + 1) << 5,
                        buf ? a0 : a1, buf ? b0 : b1, tid);
            wait_async6();
        } else {
            wait_async0();
        }
        __syncthreads();

        const char* la = lAb[buf];
        const char* lb = lBb[buf];
        FragU fb[2];
        for (int tj = 0; tj < 2; ++tj) {
            const uint4* p = (const uint4*)(lb + (wcol * 32 + tj * 16 + l15) * ROWP);
            fb[tj].u[0] = p[half];
            fb[tj].u[1] = p[2 + half];
        }
        for (int ti = 0; ti < 4; ++ti) {
            FragU fa;
            const uint4* p = (const uint4*)(la + (wrow * 64 + ti * 16 + l15) * ROWP);
            fa.u[0] = p[half];
            fa.u[1] = p[2 + half];
            acc[ti][0] = __builtin_amdgcn_wmma_f32_16x16x32_bf16(
                false, fa.f, false, fb[0].f, (short)0, acc[ti][0], false, false);
            acc[ti][1] = __builtin_amdgcn_wmma_f32_16x16x32_bf16(
                false, fa.f, false, fb[1].f, (short)0, acc[ti][1], false, false);
        }
        __syncthreads();
    }

    // epilogue: C/D layout -> m = v + 8*half, n = lane&15
    for (int ti = 0; ti < 4; ++ti) {
        for (int tj = 0; tj < 2; ++tj) {
            int n = n0 + wcol * 32 + tj * 16 + l15;
            int mb = m0 + wrow * 64 + ti * 16 + 8 * half;
            float bv = bias[n];
            for (int v = 0; v < 8; ++v)
                C[(size_t)(mb + v) * N + n] = acc[ti][tj][v] + bv;
        }
    }
}

// ---------------------------------------------------------------------------
// BatchNorm stats over batch (per UE, per feature), then BN+mish -> bf16
// ---------------------------------------------------------------------------
__global__ void k_bnstats(const float* __restrict__ x, float* __restrict__ mean,
                          float* __restrict__ var, int F) {
    int uf = blockIdx.x;
    int u = uf / F, f = uf % F;
    int tid = threadIdx.x;
    float s1 = 0.f, s2 = 0.f;
    for (int b = tid; b < CB; b += 256) {
        float v = x[((size_t)u * CB + b) * F + f];
        s1 += v; s2 += v * v;
    }
    __shared__ float r1[256], r2[256];
    r1[tid] = s1; r2[tid] = s2;
    __syncthreads();
    for (int o = 128; o > 0; o >>= 1) {
        if (tid < o) { r1[tid] += r1[tid + o]; r2[tid] += r2[tid + o]; }
        __syncthreads();
    }
    if (tid == 0) {
        float m = r1[0] * (1.f / CB);
        mean[uf] = m;
        var[uf] = r2[0] * (1.f / CB) - m * m;
    }
}

__global__ void k_bnmish(const float* __restrict__ x, const float* __restrict__ mean,
                         const float* __restrict__ var, const float* __restrict__ g,
                         const float* __restrict__ bb, unsigned short* __restrict__ o, int F) {
    size_t i = (size_t)blockIdx.x * 256 + threadIdx.x;
    if (i >= (size_t)CM * F) return;
    int f = (int)(i % F);
    size_t r = i / F;
    int u = (int)(r >> 12);
    int uf = u * F + f;
    float y = g[f] * (x[i] - mean[uf]) * rsqrtf(var[uf] + 1e-5f) + bb[f];
    o[i] = f2bf(mishf(y));
}

// ---------------------------------------------------------------------------
// Residual VQ: one wave per row, 8 codewords per lane, straight-through value=q
// ---------------------------------------------------------------------------
__global__ void k_vq(const float* __restrict__ enc, const float* __restrict__ cb,
                     unsigned short* __restrict__ vqout, float* __restrict__ loss) {
    __shared__ float lr[8][64];
    __shared__ float lq[8][64];
    int tid = threadIdx.x, wave = tid >> 5, lane = tid & 31;
    int row = blockIdx.x * 8 + wave;
    int k0 = lane * 2;
    lr[wave][k0]     = enc[(size_t)row * CD + k0];
    lr[wave][k0 + 1] = enc[(size_t)row * CD + k0 + 1];
    lq[wave][k0] = 0.f; lq[wave][k0 + 1] = 0.f;
    __syncthreads();

    for (int nq = 0; nq < CNQ; ++nq) {
        float best = 3.4e38f; int bi = 0;
        for (int c8 = 0; c8 < 8; ++c8) {
            int c = lane * 8 + c8;
            const float* cw = cb + ((size_t)nq * CK + c) * CD;
            float d = 0.f;
            for (int k = 0; k < CD; ++k) { float t = lr[wave][k] - cw[k]; d += t * t; }
            if (d < best) { best = d; bi = c; }
        }
        for (int o = 16; o > 0; o >>= 1) {
            float ob = __shfl_xor(best, o, 32);
            int   oi = __shfl_xor(bi, o, 32);
            if (ob < best || (ob == best && oi < bi)) { best = ob; bi = oi; }
        }
        const float* cw = cb + ((size_t)nq * CK + bi) * CD;
        lr[wave][k0]     -= cw[k0];     lq[wave][k0]     += cw[k0];
        lr[wave][k0 + 1] -= cw[k0 + 1]; lq[wave][k0 + 1] += cw[k0 + 1];
        __syncthreads();
    }
    vqout[(size_t)row * CD + k0]     = f2bf(lq[wave][k0]);
    vqout[(size_t)row * CD + k0 + 1] = f2bf(lq[wave][k0 + 1]);
    float part = lr[wave][k0] * lr[wave][k0] + lr[wave][k0 + 1] * lr[wave][k0 + 1];
    for (int o = 16; o > 0; o >>= 1) part += __shfl_xor(part, o, 32);
    if (lane == 0) atomicAdd(loss, part * (1.f / ((float)CM * CD)));
}

// ---------------------------------------------------------------------------
// GNN combine: y = mish(m1[u] + m4[u]*sum_{j!=u}(m2[j]*m3[j])/3) -> bf16
// ---------------------------------------------------------------------------
__global__ void k_gnn(const float* __restrict__ M1, const float* __restrict__ M2,
                      const float* __restrict__ M3, const float* __restrict__ M4,
                      unsigned short* __restrict__ o) {
    size_t i = (size_t)blockIdx.x * 256 + threadIdx.x;   // U*B*512
    if (i >= (size_t)CM * 512) return;
    int f = (int)(i & 511);
    size_t r = i >> 9;
    int u = (int)(r >> 12);
    int b = (int)(r & 4095);
    const size_t stride = (size_t)CB * 512;
    size_t base = (size_t)b * 512 + f;
    float s = 0.f, own = 0.f;
    for (int j = 0; j < CU; ++j) {
        float p = M2[j * stride + base] * M3[j * stride + base];
        s += p;
        if (j == u) own = p;
    }
    float agg = M4[u * stride + base] * (s - own) * (1.f / 3.f);
    o[i] = f2bf(mishf(M1[u * stride + base] + agg));
}

// ---------------------------------------------------------------------------
// Final: precoder build + power normalize + W_ri out + SINR/rates accumulation
// ---------------------------------------------------------------------------
__global__ void k_final(const float* __restrict__ mo, const float* __restrict__ xr,
                        const float* __restrict__ xi, float* __restrict__ out) {
    int b = blockIdx.x, tid = threadIdx.x;
    int a = tid >> 2, u = tid & 3;
    __shared__ float red[256];
    __shared__ float lWr[64][4], lWi[64][4];
    __shared__ float p2s[16];

    size_t mrow = ((size_t)u * CB + b) * (2 * CA);
    float wre = mo[mrow + CA + a];     // real = mo[..., A:]
    float wim = mo[mrow + a];          // imag = mo[..., :A]
    red[tid] = wre * wre + wim * wim;
    __syncthreads();
    for (int o = 128; o > 0; o >>= 1) {
        if (tid < o) red[tid] += red[tid + o];
        __syncthreads();
    }
    float s = rsqrtf(red[0]);
    wre *= s; wim *= s;
    out[((size_t)b * 128 + a) * 4 + u]      = wre;  // W_ri[b, a,    u]
    out[((size_t)b * 128 + 64 + a) * 4 + u] = wim;  // W_ri[b, A+a,  u]
    lWr[a][u] = wre; lWi[a][u] = wim;
    __syncthreads();

    if (tid < 16) {
        int uu = tid >> 2, vv = tid & 3;
        const float* hr = xr + ((size_t)b * CU + uu) * CA;
        const float* hi = xi + ((size_t)b * CU + uu) * CA;
        float re = 0.f, im = 0.f;
        for (int aa = 0; aa < CA; ++aa) {
            float cr = lWr[aa][vv], ci = lWi[aa][vv];
            re += hr[aa] * cr - hi[aa] * ci;
            im += hr[aa] * ci + hi[aa] * cr;
        }
        p2s[tid] = re * re + im * im;
    }
    __syncthreads();
    if (tid < 4) {
        float P = p2s[tid * 4 + tid];
        float I = 0.f;
        for (int v = 0; v < 4; ++v) I += p2s[tid * 4 + v];
        I -= P;
        float sinr = P / (0.1f + I);           // n_power = 10^(-SNR/10) = 0.1
        float rate = log2f(1.f + sinr) * (1.f / CB);
        atomicAdd(&out[2097153 + tid], rate);
        atomicAdd(&out[2097157], rate);
    }
}

// ---------------------------------------------------------------------------
// Host-side orchestration (graph-capture safe: only launches on `stream`)
// ---------------------------------------------------------------------------
extern "C" void kernel_launch(void* const* d_in, const int* in_sizes, int n_in,
                              void* d_out, int out_size, void* d_ws, size_t ws_size,
                              hipStream_t stream) {
    (void)in_sizes; (void)n_in; (void)out_size; (void)ws_size;
    // pytree (sorted-key) flatten order of setup_inputs()
    enum {
        IN_NOISE = 0, IN_CB = 1,
        IN_BN1B = 2, IN_BN1G = 3, IN_BN2B = 4, IN_BN2G = 5, IN_BN3B = 6, IN_BN3G = 7,
        IN_L1B = 8, IN_L1W = 9, IN_L2B = 10, IN_L2W = 11, IN_L3B = 12, IN_L3W = 13,
        IN_L4B = 14, IN_L4W = 15,
        IN_G1M1B = 16, IN_G1M1W = 17, IN_G1M2B = 18, IN_G1M2W = 19,
        IN_G1M3B = 20, IN_G1M3W = 21, IN_G1M4B = 22, IN_G1M4W = 23,
        IN_G2M1B = 24, IN_G2M1W = 25, IN_G2M2B = 26, IN_G2M2W = 27,
        IN_G2M3B = 28, IN_G2M3W = 29, IN_G2M4B = 30, IN_G2M4W = 31,
        IN_INIT1B = 32, IN_INIT1W = 33, IN_INIT2B = 34, IN_INIT2W = 35,
        IN_OUT1B = 36, IN_OUT1W = 37, IN_OUT2B = 38, IN_OUT2W = 39,
        IN_WPI = 40, IN_WPR = 41, IN_XI = 42, IN_XR = 43
    };
    #define FP(i) ((const float*)d_in[i])

    char* ws = (char*)d_ws;
    float* out = (float*)d_out;

    // workspace arena
    unsigned short* wbf   = (unsigned short*)ws;                        // bf16 weights (<8MB)
    float* norms          = (float*)(ws + ((size_t)8 << 20));
    float* meanv          = norms + 64;
    float* varv           = meanv + 4096;
    float* encf           = varv + 4096;                                // 16384*64 f32
    unsigned short* vqb   = (unsigned short*)(encf + (size_t)CM * CD);  // 16384*64 bf16
    unsigned short* pcat  = vqb + (size_t)CM * CD;                      // 16384*128 bf16
    float* buf0           = (float*)(ws + ((size_t)32 << 20));          // 64MB f32
    float* buf1           = (float*)(ws + ((size_t)96 << 20));          // 64MB f32
    unsigned short* actA  = (unsigned short*)(ws + ((size_t)160 << 20));// 32MB bf16
    unsigned short* actB  = (unsigned short*)(ws + ((size_t)192 << 20));// 32MB bf16

    // bf16 transposed weight slots [N,K]
    unsigned short* enc1w  = wbf;
    unsigned short* enc2w  = enc1w + 131072;
    unsigned short* enc3w  = enc2w + 524288;
    unsigned short* enc4w  = enc3w + 131072;
    unsigned short* init1w = enc4w + 16384;
    unsigned short* init2w = init1w + 65536;
    unsigned short* g1w0   = init2w + 524288;
    unsigned short* g2w0   = g1w0 + 4 * 262144;
    unsigned short* out1w  = g2w0 + 4 * 262144;
    unsigned short* out2w  = out1w + 131072;

    k_zero_tail<<<1, 32, 0, stream>>>(out);

    // weight fp32 [K,N] -> bf16 transposed [N,K]
    auto cvt = [&](int idx, unsigned short* dst, int N, int K) {
        k_f2bf_t<<<(N * K + 255) / 256, 256, 0, stream>>>(FP(idx), dst, N, K);
    };
    cvt(IN_L1W, enc1w, 1024, 128);
    cvt(IN_L2W, enc2w, 512, 1024);
    cvt(IN_L3W, enc3w, 256, 512);
    cvt(IN_L4W, enc4w, 64, 256);
    cvt(IN_INIT1W, init1w, 1024, 64);
    cvt(IN_INIT2W, init2w, 512, 1024);
    for (int m = 0; m < 4; ++m) {
        cvt(IN_G1M1W + 2 * m, g1w0 + (size_t)m * 262144, 512, 512);
        cvt(IN_G2M1W + 2 * m, g2w0 + (size_t)m * 262144, 512, 512);
    }
    cvt(IN_OUT1W, out1w, 256, 512);
    cvt(IN_OUT2W, out2w, 128, 256);

    // pilot
    k_pilot_norm<<<1, 64, 0, stream>>>(FP(IN_WPR), FP(IN_WPI), norms);
    k_pilot<<<(CM * CT) / 256, 256, 0, stream>>>(FP(IN_XR), FP(IN_XI), FP(IN_WPR),
                                                 FP(IN_WPI), norms, FP(IN_NOISE), pcat);

    auto gemm = [&](const unsigned short* Aa, const unsigned short* Ww,
                    const float* bias, float* Cc, int N, int K) {
        dim3 g(N / 64, CM / 128);
        k_gemm<<<g, 128, 0, stream>>>(Aa, Ww, bias, Cc, N, K);
    };
    auto bn = [&](const float* pre, const float* gg, const float* bb,
                  unsigned short* act, int Fd) {
        k_bnstats<<<CU * Fd, 256, 0, stream>>>(pre, meanv, varv, Fd);
        k_bnmish<<<(int)(((size_t)CM * Fd) / 256), 256, 0, stream>>>(pre, meanv, varv, gg,
                                                                     bb, act, Fd);
    };
    auto mishcvt = [&](const float* pre, unsigned short* act, int Fd) {
        k_mish_bf<<<(int)(((size_t)CM * Fd) / 256), 256, 0, stream>>>(pre, act,
                                                                      (size_t)CM * Fd);
    };

    // encoder (BN is per-UE via row grouping [U,B,.])
    gemm(pcat, enc1w, FP(IN_L1B), buf0, 1024, 128);
    bn(buf0, FP(IN_BN1G), FP(IN_BN1B), actA, 1024);
    gemm(actA, enc2w, FP(IN_L2B), buf1, 512, 1024);
    bn(buf1, FP(IN_BN2G), FP(IN_BN2B), actB, 512);
    gemm(actB, enc3w, FP(IN_L3B), buf0, 256, 512);
    bn(buf0, FP(IN_BN3G), FP(IN_BN3B), actA, 256);
    gemm(actA, enc4w, FP(IN_L4B), buf1, 64, 256);
    k_tanh_f32<<<(CM * CD) / 256, 256, 0, stream>>>(buf1, encf, (size_t)CM * CD);

    // residual VQ + loss
    k_vq<<<CM / 8, 256, 0, stream>>>(encf, FP(IN_CB), vqb, out + 2097152);

    // init MLP
    gemm(vqb, init1w, FP(IN_INIT1B), buf0, 1024, 64);
    mishcvt(buf0, actA, 1024);
    gemm(actA, init2w, FP(IN_INIT2B), buf1, 512, 1024);
    mishcvt(buf1, actB, 512);

    // GNN layers
    float* M1 = buf0;
    float* M2 = buf0 + (size_t)CM * 512;
    float* M3 = buf1;
    float* M4 = buf1 + (size_t)CM * 512;
    const int gnn_grid = (int)(((size_t)CM * 512) / 256);

    gemm(actB, g1w0 + 0 * 262144, FP(IN_G1M1B), M1, 512, 512);
    gemm(actB, g1w0 + 1 * 262144, FP(IN_G1M2B), M2, 512, 512);
    gemm(actB, g1w0 + 2 * 262144, FP(IN_G1M3B), M3, 512, 512);
    gemm(actB, g1w0 + 3 * 262144, FP(IN_G1M4B), M4, 512, 512);
    k_gnn<<<gnn_grid, 256, 0, stream>>>(M1, M2, M3, M4, actA);

    gemm(actA, g2w0 + 0 * 262144, FP(IN_G2M1B), M1, 512, 512);
    gemm(actA, g2w0 + 1 * 262144, FP(IN_G2M2B), M2, 512, 512);
    gemm(actA, g2w0 + 2 * 262144, FP(IN_G2M3B), M3, 512, 512);
    gemm(actA, g2w0 + 3 * 262144, FP(IN_G2M4B), M4, 512, 512);
    k_gnn<<<gnn_grid, 256, 0, stream>>>(M1, M2, M3, M4, actB);

    // output MLP
    gemm(actB, out1w, FP(IN_OUT1B), buf0, 256, 512);
    mishcvt(buf0, actA, 256);
    gemm(actA, out2w, FP(IN_OUT2B), buf1, 128, 256);   // mo fp32 [U*B, 128]

    // precoder normalize + W_ri + rates
    k_final<<<CB, 256, 0, stream>>>(buf1, FP(IN_XR), FP(IN_XI), out);
    #undef FP
}
